// StaticRefinerTuner_15616501088912
// MI455X (gfx1250) — matrix operand order repaired and checked by reading, not compiled, not run on hardware
//
#include <hip/hip_runtime.h>

typedef __attribute__((ext_vector_type(16))) _Float16 v16h;
typedef __attribute__((ext_vector_type(8)))  float    v8f;

#define KSZ   27          // gaussian kernel size (sigma=4 -> 27, odd)
#define HOFF  14          // output row = label_x + i - 14  (i in [0,26])
#define NLAB  1024        // labels per image
#define CAP   224         // max compacted labels per 16x128 strip
#define IMG   768         // H == W == 768
#define TILES_1D 48       // 768 / 16
#define WPB   8           // waves per block (256 threads, wave32)

__global__ __launch_bounds__(256)
void gauss_splat_wmma(const int* __restrict__ labels,   // [B, NLAB, 2] (x=row, y=col)
                      const float* __restrict__ sig,    // [1]
                      float* __restrict__ out)          // [B, 768, 768]
{
    __shared__ __align__(16) int slab[NLAB * 2];        // staged labels (8 KB)
    __shared__ float    gtmp[KSZ];
    __shared__ _Float16 gh[32];                         // normalized 1-D gaussian; [27..31]=0
    __shared__ int      slx[CAP + 32];                  // compacted rows (+pad margin)
    __shared__ int      sly[CAP + 32];                  // compacted cols (+pad margin)
    __shared__ int      scount;

    const int tid  = threadIdx.x;
    const int lane = tid & 31;
    const int wave = tid >> 5;
    const int h    = lane >> 4;      // lane half (0/1)
    const int m    = lane & 15;      // row (A) / col (B,D) within tile
    const int b    = blockIdx.y;

    const int* lab = labels + (size_t)b * NLAB * 2;

    // ---- async-stage labels into LDS (ASYNCcnt path), 16 B per lane per iter ----
    {
        const int base = tid * 4;                       // dword index
        #pragma unroll
        for (int it = 0; it < 2; ++it) {
            const int di = base + it * 1024;            // 256 thr * 4 dwords
            unsigned lds_off = (unsigned)(size_t)(&slab[di]);
            unsigned long long ga = (unsigned long long)(size_t)(lab + di);
            asm volatile("global_load_async_to_lds_b128 %0, %1, off"
                         :: "v"(lds_off), "v"(ga) : "memory");
        }
    }
    if (tid == 0) scount = 0;
    if (tid < KSZ) {                                    // raw 1-D gaussian
        float d = (float)(tid - KSZ / 2);
        float s = sig[0];
        gtmp[tid] = __expf(-d * d / (2.0f * s * s));
    }
    asm volatile("s_wait_asynccnt 0x0" ::: "memory");
    __syncthreads();

    // ---- normalized f16 table, zero-padded to 32 so a clamped index is branch-free ----
    if (tid < 32) {
        float ssum = 0.f;
        #pragma unroll
        for (int i = 0; i < KSZ; ++i) ssum += gtmp[i];
        gh[tid] = (_Float16)((tid < KSZ) ? (gtmp[tid] / ssum) : 0.f);
    }

    // ---- strip owned by this block: rows [R0,R0+15], cols [CS0,CS0+127] ----
    const int t0  = blockIdx.x * WPB;                   // first tile of strip
    const int R0  = (t0 / TILES_1D) * 16;
    const int CS0 = (t0 % TILES_1D) * 16;
    const int C0  = CS0 + wave * 16;                    // this wave's tile columns

    // ---- cooperative compaction: each wave scans 128 labels vs the strip window ----
    for (int g = 0; g < 4; ++g) {
        const int idx = wave * 128 + g * 32 + lane;
        const int x = slab[2 * idx + 0];
        const int y = slab[2 * idx + 1];
        const bool hit = (x - HOFF <= R0 + 15)  && (x + (KSZ - 1 - HOFF) >= R0) &&
                         (y - HOFF <= CS0 + 127) && (y + (KSZ - 1 - HOFF) >= CS0);
        const unsigned mask = (unsigned)__ballot(hit);  // wave32: low 32 bits
        const int cnt = __popc(mask);
        int wbase = 0;
        if (lane == 0 && cnt) wbase = atomicAdd(&scount, cnt);
        wbase = __shfl(wbase, 0);
        const int pos = wbase + __popc(mask & ((1u << lane) - 1u));
        if (hit && pos < CAP) { slx[pos] = x; sly[pos] = y; }
    }
    __syncthreads();

    // ---- pad list to a multiple of 32 with sentinels -> unconditional chunk reads ----
    int L = scount; if (L > CAP) L = CAP;
    const int Lp = (L + 31) & ~31;
    if (tid < 32) {
        const int p = L + tid;
        if (p < Lp) { slx[p] = -100000; sly[p] = -100000; }
    }
    __syncthreads();

    // ---- D(16x16) = sum_k g_row(k) (x) g_col(k) via v_wmma_f32_16x16x32_f16 ----
    v8f acc = {};
    const int rA = R0 + m + HOFF;                       // gh index = rA - x
    const int cB = C0 + m + HOFF;                       // gh index = cB - y
    for (int c0 = 0; c0 < Lp; c0 += 32) {
        v16h A, B;
        #pragma unroll
        for (int e = 0; e < 16; ++e) {
            // A 16x32 f16 layout: lane half 0 -> K={0..7,16..23}, half 1 -> +8
            const int ka = ((e < 8) ? e : e + 8) + 8 * h;
            // B 32x16 f16 layout: lane half 0 -> K={0..7,16..23}, half 1 -> +8
            const int kb = ((e < 8) ? e : e + 8) + 8 * h;
            unsigned da = (unsigned)(rA - slx[c0 + ka]);
            unsigned db = (unsigned)(cB - sly[c0 + kb]);
            da = (da > 31u) ? 31u : da;                 // clamp -> gh[27..31]==0
            db = (db > 31u) ? 31u : db;
            A[e] = gh[da];
            B[e] = gh[db];
        }
        acc = __builtin_amdgcn_wmma_f32_16x16x32_f16(
            /*neg_a=*/false, A, /*neg_b=*/false, B,
            /*c_mod=*/(short)0, acc, /*reuse_a=*/false, /*reuse_b=*/false);
    }

    // ---- store D: VGPR v -> row (v + 8*h), col = m ; coalesced, full coverage ----
    float* orow = out + ((size_t)b * IMG + R0) * IMG + C0 + m;
    #pragma unroll
    for (int v = 0; v < 8; ++v) {
        orow[(size_t)(v + 8 * h) * IMG] = acc[v];
    }
}

extern "C" void kernel_launch(void* const* d_in, const int* in_sizes, int n_in,
                              void* d_out, int out_size, void* d_ws, size_t ws_size,
                              hipStream_t stream) {
    // inputs: d_in[0] = batch_images (unused by reference math),
    //         d_in[1] = batch_labels int32 [B, 1024, 2],
    //         d_in[2] = sigma float [1]
    const int*   labels = (const int*)d_in[1];
    const float* sig    = (const float*)d_in[2];
    float*       out    = (float*)d_out;

    const int B = in_sizes[1] / (NLAB * 2);             // 32
    dim3 grid((TILES_1D * TILES_1D) / WPB, B);          // 288 x 32 blocks
    gauss_splat_wmma<<<grid, 256, 0, stream>>>(labels, sig, out);
}
